// PointNet2SSGCls_20916490731530
// MI455X (gfx1250) — matrix operand order repaired and checked by reading, not compiled, not run on hardware
//
#include <hip/hip_runtime.h>
#include <hip/hip_bf16.h>

// ---------------------------------------------------------------------------
// PointNet++ SSG classifier forward for MI455X (gfx1250, wave32, WMMA).
// GEMMs run on v_wmma_f32_16x16x32_f16 with activations/weights stored in
// fragment-linear (WMMA-native) layout so the inner loop is pure
// global_load_b128 -> v_wmma, no LDS staging, no tail branches.
// ---------------------------------------------------------------------------

typedef _Float16 v16h __attribute__((ext_vector_type(16)));
typedef float    v8f  __attribute__((ext_vector_type(8)));

#define BN_EPS 1e-5f

// Fragment-linear X layout: element (c, p) of the [Cpad, P] activation matrix
// lives at Xf[(pt*nKc + kcI)*512 + lane*16 + e] where, per the CDNA5 B-matrix
// VGPR striping for 16-bit 32x16 fragments:
//   pt = p/16, n = p%16, kcI = c/32, kk = c%32, half = (kk>>3)&1,
//   e = (kk&7) | ((kk&16)>>1), lane = half*16 + n.
__device__ __forceinline__ size_t pn2_xf(int c, size_t p, int nKc) {
  size_t pt = p >> 4;
  int n    = (int)(p & 15);
  int kcI  = c >> 5;
  int kk   = c & 31;
  int half = (kk >> 3) & 1;
  int e    = (kk & 7) | ((kk & 16) >> 1);
  return (pt * (size_t)nKc + kcI) * 512 + (size_t)(half * 16 + n) * 16 + e;
}

// ---------------------------------------------------------------------------
// points [B,3,N] -> xyz [B,N,3]
// ---------------------------------------------------------------------------
__global__ void pn2_pts_transpose(const float* __restrict__ pts,
                                  float* __restrict__ xyz, int B, int N) {
  int idx = blockIdx.x * blockDim.x + threadIdx.x;
  if (idx >= B * N * 3) return;
  int j = idx % 3;
  int n = (idx / 3) % N;
  int b = idx / (3 * N);
  xyz[idx] = pts[((size_t)b * 3 + j) * (size_t)N + n];
}

// ---------------------------------------------------------------------------
// Farthest point sampling: one 256-thread block per batch element.
// ---------------------------------------------------------------------------
template <int PPT>
__global__ __launch_bounds__(256) void pn2_fps(const float* __restrict__ xyz, // [B,N,3]
                                               int N, int m,
                                               int* __restrict__ cidx,        // [B,m]
                                               float* __restrict__ nxyz) {    // [B,m,3]
  const int b = blockIdx.x;
  const int tid = threadIdx.x;
  __shared__ float sd[256];
  __shared__ int   si[256];
  const float* base = xyz + (size_t)b * N * 3;
  float mind[PPT];
#pragma unroll
  for (int i = 0; i < PPT; ++i) mind[i] = 3.4e38f;
  int far = 0;
  for (int it = 0; it < m; ++it) {
    if (tid == 0) {
      cidx[b * m + it] = far;
      nxyz[((size_t)b * m + it) * 3 + 0] = base[far * 3 + 0];
      nxyz[((size_t)b * m + it) * 3 + 1] = base[far * 3 + 1];
      nxyz[((size_t)b * m + it) * 3 + 2] = base[far * 3 + 2];
    }
    float cx = base[far * 3 + 0], cy = base[far * 3 + 1], cz = base[far * 3 + 2];
    float bestd = -1.0f;
    int   besti = 0;
#pragma unroll
    for (int i = 0; i < PPT; ++i) {
      int n = i * 256 + tid;
      float dx = base[n * 3 + 0] - cx;
      float dy = base[n * 3 + 1] - cy;
      float dz = base[n * 3 + 2] - cz;
      float d  = dx * dx + dy * dy + dz * dz;
      float md = fminf(mind[i], d);
      mind[i]  = md;
      if (md > bestd) { bestd = md; besti = n; }
    }
    sd[tid] = bestd; si[tid] = besti;
    __syncthreads();
    for (int s = 128; s > 0; s >>= 1) {
      if (tid < s && sd[tid + s] > sd[tid]) { sd[tid] = sd[tid + s]; si[tid] = si[tid + s]; }
      __syncthreads();
    }
    far = si[0];
    __syncthreads();
  }
}

// ---------------------------------------------------------------------------
// Ball query + group: one wave32 per centroid (8 / block). First K in-radius
// in index order via wave prefix sum; pad with first hit; write swizzled f16.
// ---------------------------------------------------------------------------
__global__ __launch_bounds__(256) void pn2_group(
    const float* __restrict__ xyz,   // [B,N,3]
    const float* __restrict__ nxyz,  // [B,M,3]
    const float* __restrict__ feat,  // [B,N,Cf] or nullptr
    _Float16* __restrict__ Xf,       // swizzled [Cpad, B*M*K]
    int B, int N, int M, int K, int Cf, int nKc, float r2) {
  __shared__ int nidAll[8 * 64];
  const int wslot = threadIdx.x >> 5;
  const int lane  = threadIdx.x & 31;
  int* nid = nidAll + wslot * 64;
  const int wid = blockIdx.x * 8 + wslot;
  const int tot = B * M;
  const int chunk = N / 32;
  float cx = 0.f, cy = 0.f, cz = 0.f;
  const float* bx = xyz;
  int total = 0, cnt = 0, exc = 0, b = 0;
  if (wid < tot) {
    b  = wid / M;
    cx = nxyz[(size_t)wid * 3 + 0];
    cy = nxyz[(size_t)wid * 3 + 1];
    cz = nxyz[(size_t)wid * 3 + 2];
    bx = xyz + (size_t)b * N * 3;
    const int s0 = lane * chunk;
    for (int j = 0; j < chunk; ++j) {
      int n = s0 + j;
      float dx = bx[n * 3 + 0] - cx, dy = bx[n * 3 + 1] - cy, dz = bx[n * 3 + 2] - cz;
      if (dx * dx + dy * dy + dz * dz <= r2) ++cnt;
    }
    int inc = cnt;
    for (int off = 1; off < 32; off <<= 1) {
      int v = __shfl_up(inc, off, 32);
      if (lane >= off) inc += v;
    }
    exc   = inc - cnt;
    total = __shfl(inc, 31, 32);
    if (exc < K && cnt > 0) {
      int wpos = exc;
      for (int j = 0; j < chunk && wpos < K; ++j) {
        int n = s0 + j;
        float dx = bx[n * 3 + 0] - cx, dy = bx[n * 3 + 1] - cy, dz = bx[n * 3 + 2] - cz;
        if (dx * dx + dy * dy + dz * dz <= r2) { nid[wpos] = n; ++wpos; }
      }
    }
  }
  __syncthreads();
  if (wid < tot) {
    int t = total < K ? total : K;   // t >= 1: centroid itself is in radius
    for (int j = t + lane; j < K; j += 32) nid[j] = nid[0];
  }
  __syncthreads();
  if (wid < tot) {
    for (int ki = lane; ki < K; ki += 32) {
      int n = nid[ki];
      size_t p = (size_t)wid * K + ki;
      Xf[pn2_xf(0, p, nKc)] = (_Float16)(bx[n * 3 + 0] - cx);
      Xf[pn2_xf(1, p, nKc)] = (_Float16)(bx[n * 3 + 1] - cy);
      Xf[pn2_xf(2, p, nKc)] = (_Float16)(bx[n * 3 + 2] - cz);
    }
    if (Cf > 0) {
      const float* bf = feat + (size_t)b * N * Cf;
      for (int t2 = lane; t2 < K * Cf; t2 += 32) {
        int ki = t2 % K;
        int c  = t2 / K;
        int n  = nid[ki];
        size_t p = (size_t)wid * K + ki;
        Xf[pn2_xf(3 + c, p, nKc)] = (_Float16)bf[(size_t)n * Cf + c];
      }
    }
  }
}

// ---------------------------------------------------------------------------
// Zero-fill f16 range (pads of the swizzled buffers).
// ---------------------------------------------------------------------------
__global__ void pn2_zero_f16(_Float16* __restrict__ p, size_t count) {
  size_t idx = (size_t)blockIdx.x * blockDim.x + threadIdx.x;
  if (idx < count) p[idx] = (_Float16)0.f;
}

// ---------------------------------------------------------------------------
// Weights fp32 [O,C] -> f16 A-fragment-linear layout, zero-padded to
// [Opad, Cpad].  A-matrix striping (16-bit 16x32): lane = half*16 + (o%16),
// element e holds K = ((e>>3)<<4) + half*8 + (e&7).
// ---------------------------------------------------------------------------
__global__ void pn2_conv_w(const float* __restrict__ W, _Float16* __restrict__ Wf,
                           int O, int C, int Opad, int Cpad) {
  int idx = blockIdx.x * blockDim.x + threadIdx.x;
  if (idx >= Opad * Cpad) return;
  int o = idx / Cpad;
  int c = idx % Cpad;
  float v = (o < O && c < C) ? W[(size_t)o * C + c] : 0.f;
  int nKc  = Cpad >> 5;
  int ot = o >> 4, m = o & 15;
  int kcI = c >> 5, kk = c & 31;
  int halfA = (kk >> 3) & 1;
  int e = (kk & 7) | ((kk & 16) >> 1);
  size_t dst = ((size_t)ot * nKc + kcI) * 512 + (size_t)(halfA * 16 + m) * 16 + e;
  Wf[dst] = (_Float16)v;
}

// ---------------------------------------------------------------------------
// WMMA GEMM: Y[O,P] fp32 (+bias) = Wf x Xf, both fragment-linear f16.
// 256 threads = 8 waves (2x4); each wave owns a 32(O) x 32(P) macro-tile:
// 4 independent accumulators, 4 v16h coalesced loads + 4 WMMA per K-chunk.
// ---------------------------------------------------------------------------
__global__ __launch_bounds__(256) void pn2_gemm_f16(
    const _Float16* __restrict__ Wf,  // [Opad/16][nKc][512]
    const _Float16* __restrict__ Xf,  // [Ppad/16][nKc][512]
    const float* __restrict__ bias,   // [O] or nullptr
    float* __restrict__ Y,            // [O, P]
    int O, int nKc, int P) {
  const int tid  = threadIdx.x;
  const int wave = tid >> 5;
  const int lane = tid & 31;
  const int ow = wave >> 2;                 // 0..1
  const int pw = wave & 3;                  // 0..3
  const int ot0 = blockIdx.y * 4 + ow * 2;  // two O tiles
  const int pt0 = blockIdx.x * 8 + pw * 2;  // two P tiles
  const size_t tileStride = (size_t)nKc * 512;
  const _Float16* aP = Wf + (size_t)ot0 * tileStride + lane * 16;
  const _Float16* bP = Xf + (size_t)pt0 * tileStride + lane * 16;

  v8f acc00 = {}, acc01 = {}, acc10 = {}, acc11 = {};
  for (int kc = 0; kc < nKc; ++kc) {
    const size_t koff = (size_t)kc * 512;
    v16h a0 = *(const v16h*)(aP + koff);
    v16h a1 = *(const v16h*)(aP + tileStride + koff);
    v16h b0 = *(const v16h*)(bP + koff);
    v16h b1 = *(const v16h*)(bP + tileStride + koff);
    __builtin_prefetch(bP + koff + 512, 0, 1);
    __builtin_prefetch(bP + tileStride + koff + 512, 0, 1);
    acc00 = __builtin_amdgcn_wmma_f32_16x16x32_f16(false, a0, false, b0, (short)0, acc00, false, false);
    acc01 = __builtin_amdgcn_wmma_f32_16x16x32_f16(false, a0, false, b1, (short)0, acc01, false, false);
    acc10 = __builtin_amdgcn_wmma_f32_16x16x32_f16(false, a1, false, b0, (short)0, acc10, false, false);
    acc11 = __builtin_amdgcn_wmma_f32_16x16x32_f16(false, a1, false, b1, (short)0, acc11, false, false);
  }
  // C/D layout: VGPR r -> M = r + 8*(lane>=16), N = lane&15
  const int n    = lane & 15;
  const int half = lane >> 4;
  auto store = [&](const v8f& acc, int ot, int pt) {
    int p = pt * 16 + n;
    if (p >= P) return;
#pragma unroll
    for (int r = 0; r < 8; ++r) {
      int o = ot * 16 + half * 8 + r;
      if (o < O) {
        float v = acc[r];
        if (bias) v += bias[o];
        Y[(size_t)o * P + p] = v;
      }
    }
  };
  store(acc00, ot0, pt0);
  store(acc01, ot0, pt0 + 1);
  store(acc10, ot0 + 1, pt0);
  store(acc11, ot0 + 1, pt0 + 1);
}

// ---------------------------------------------------------------------------
// Per-channel training-mode BN stats over P, folded into scale/shift.
// ---------------------------------------------------------------------------
__global__ __launch_bounds__(256) void pn2_bn_stats(
    const float* __restrict__ Y, int P,
    const float* __restrict__ gamma, const float* __restrict__ beta,
    float* __restrict__ scale, float* __restrict__ shift) {
  const int c   = blockIdx.x;
  const int tid = threadIdx.x;
  const float* row = Y + (size_t)c * P;
  float s = 0.f, sq = 0.f;
  for (int p = tid; p < P; p += 256) {
    float v = row[p];
    s += v; sq += v * v;
  }
  __shared__ float ss[256], sv[256];
  ss[tid] = s; sv[tid] = sq;
  __syncthreads();
  for (int st = 128; st > 0; st >>= 1) {
    if (tid < st) { ss[tid] += ss[tid + st]; sv[tid] += sv[tid + st]; }
    __syncthreads();
  }
  if (tid == 0) {
    float mean = ss[0] / (float)P;
    float var  = sv[0] / (float)P - mean * mean;
    float sc   = gamma[c] * rsqrtf(var + BN_EPS);
    scale[c] = sc;
    shift[c] = beta[c] - mean * sc;
  }
}

// ---------------------------------------------------------------------------
// Xf[c,p] = f16(relu(Y*scale+shift)), swizzled output.
// ---------------------------------------------------------------------------
__global__ void pn2_bn_relu_f16(const float* __restrict__ Y,
                                const float* __restrict__ scale,
                                const float* __restrict__ shift,
                                _Float16* __restrict__ Xf,
                                size_t P, size_t total, int nKc) {
  size_t idx = (size_t)blockIdx.x * blockDim.x + threadIdx.x;
  if (idx >= total) return;
  size_t c = idx / P;
  size_t p = idx % P;
  float v = fmaxf(Y[idx] * scale[c] + shift[c], 0.f);
  Xf[pn2_xf((int)c, p, nKc)] = (_Float16)v;
}

// ---------------------------------------------------------------------------
// out[g,c] = max over pool window of relu(bn(Y)).  P = NG*PS, pool innermost.
// ---------------------------------------------------------------------------
__global__ void pn2_bn_relu_maxpool(const float* __restrict__ Y,
                                    const float* __restrict__ scale,
                                    const float* __restrict__ shift,
                                    float* __restrict__ out,
                                    int C, int NG, int PS) {
  int idx = blockIdx.x * blockDim.x + threadIdx.x;
  if (idx >= NG * C) return;
  int c = idx % C;
  int g = idx / C;
  size_t P = (size_t)NG * PS;
  const float* row = Y + (size_t)c * P + (size_t)g * PS;
  float sc = scale[c], sh = shift[c];
  float mx = 0.f;  // relu floor
  for (int i = 0; i < PS; ++i) mx = fmaxf(mx, fmaxf(row[i] * sc + sh, 0.f));
  out[(size_t)g * C + c] = mx;
}

// ---------------------------------------------------------------------------
// Local-stack input: swizzled concat(new_xyz2, feat2).
// ---------------------------------------------------------------------------
__global__ void pn2_concat_local(const float* __restrict__ nxyz2,  // [NG,3]
                                 const float* __restrict__ feat2,  // [NG,Cf]
                                 _Float16* __restrict__ Xf,
                                 int NG, int Cf, int nKc) {
  int idx = blockIdx.x * blockDim.x + threadIdx.x;
  if (idx >= (3 + Cf) * NG) return;
  int c = idx / NG;
  int g = idx % NG;
  float v = (c < 3) ? nxyz2[(size_t)g * 3 + c] : feat2[(size_t)g * Cf + (c - 3)];
  Xf[pn2_xf(c, (size_t)g, nKc)] = (_Float16)v;
}

// ---------------------------------------------------------------------------
// pooled [B,C] fp32 -> swizzled Xf [C,B] f16 (FC input).
// ---------------------------------------------------------------------------
__global__ void pn2_pool_to_x(const float* __restrict__ pooled,
                              _Float16* __restrict__ Xf, int B, int C, int nKc) {
  int idx = blockIdx.x * blockDim.x + threadIdx.x;
  if (idx >= B * C) return;
  int c = idx / B;
  int b = idx % B;
  Xf[pn2_xf(c, (size_t)b, nKc)] = (_Float16)pooled[(size_t)b * C + c];
}

// ---------------------------------------------------------------------------
// Y [O,P] -> out [P,O] (classifier logits; bias fused in GEMM).
// ---------------------------------------------------------------------------
__global__ void pn2_out_write(const float* __restrict__ Y, float* __restrict__ out,
                              int O, int P) {
  int idx = blockIdx.x * blockDim.x + threadIdx.x;
  if (idx >= O * P) return;
  int o = idx % O;
  int p = idx / O;
  out[idx] = Y[(size_t)o * P + p];
}

// ===========================================================================
// Host orchestration
// ===========================================================================
extern "C" void kernel_launch(void* const* d_in, const int* in_sizes, int n_in,
                              void* d_out, int out_size, void* d_ws, size_t ws_size,
                              hipStream_t stream) {
  (void)in_sizes; (void)n_in; (void)out_size; (void)ws_size;

  constexpr int B = 32, N = 4096;
  constexpr int M1 = 512, K1 = 32;      // SA1
  constexpr int M2 = 128, K2 = 64;      // SA2
  constexpr float R1SQ = 0.2f * 0.2f, R2SQ = 0.4f * 0.4f;
  const size_t P1 = (size_t)B * M1 * K1;   // 524288
  const size_t P2 = (size_t)B * M2 * K2;   // 262144
  const size_t P3 = (size_t)B * M2;        // 4096
  const size_t PB = B;                     // FC "P" = batch

  // ----- input pointers (setup_inputs flattening order) -----
  const float* points = (const float*)d_in[0];
  const float* SA1W[3] = {(const float*)d_in[1], (const float*)d_in[4], (const float*)d_in[7]};
  const float* SA1G[3] = {(const float*)d_in[2], (const float*)d_in[5], (const float*)d_in[8]};
  const float* SA1B[3] = {(const float*)d_in[3], (const float*)d_in[6], (const float*)d_in[9]};
  const float* SA2W[3] = {(const float*)d_in[10], (const float*)d_in[13], (const float*)d_in[16]};
  const float* SA2G[3] = {(const float*)d_in[11], (const float*)d_in[14], (const float*)d_in[17]};
  const float* SA2B[3] = {(const float*)d_in[12], (const float*)d_in[15], (const float*)d_in[18]};
  const float* LW[3]   = {(const float*)d_in[19], (const float*)d_in[22], (const float*)d_in[25]};
  const float* LG[3]   = {(const float*)d_in[20], (const float*)d_in[23], (const float*)d_in[26]};
  const float* LB[3]   = {(const float*)d_in[21], (const float*)d_in[24], (const float*)d_in[27]};
  const float* GW[2]   = {(const float*)d_in[28], (const float*)d_in[32]};
  const float* GBi[2]  = {(const float*)d_in[29], (const float*)d_in[33]};
  const float* GG[2]   = {(const float*)d_in[30], (const float*)d_in[34]};
  const float* GBe[2]  = {(const float*)d_in[31], (const float*)d_in[35]};
  const float* clsW    = (const float*)d_in[36];
  const float* clsB    = (const float*)d_in[37];
  float* out = (float*)d_out;

  // ----- workspace carve-out -----
  char* ws = (char*)d_ws;
  size_t off = 0;
  auto take = [&](size_t bytes) -> char* {
    char* p = ws + off;
    off = (off + bytes + 255) & ~(size_t)255;
    return p;
  };
  float*     xyz1   = (float*)take((size_t)B * N * 3 * 4);
  float*     nxyz1  = (float*)take((size_t)B * M1 * 3 * 4);
  float*     nxyz2  = (float*)take((size_t)B * M2 * 3 * 4);
  int*       cidx1  = (int*)take((size_t)B * M1 * 4);
  int*       cidx2  = (int*)take((size_t)B * M2 * 4);
  float*     feat1  = (float*)take((size_t)B * M1 * 128 * 4);
  float*     feat2  = (float*)take((size_t)B * M2 * 256 * 4);
  float*     pooled = (float*)take((size_t)B * 1024 * 4);
  float*     scaleB = (float*)take(1024 * 4);
  float*     shiftB = (float*)take(1024 * 4);
  _Float16*  Wh     = (_Float16*)take((size_t)1024 * 1024 * 2);  // max Opad*Cpad
  _Float16*  XfA    = (_Float16*)take((size_t)160 * P2 * 2);     // >= 64*P1 too
  _Float16*  XfB    = (_Float16*)take((size_t)160 * P2 * 2);
  float*     Ybuf   = (float*)take((size_t)128 * P1 * 4);        // == 256*P2

  auto cdiv = [](size_t a, size_t b) { return (unsigned)((a + b - 1) / b); };
  auto ptile8  = [](size_t P) { return ((P / 16) + 7) & ~(size_t)7; };
  auto xfElems = [&](size_t P, int Cpad) { return ptile8(P) * (size_t)(Cpad / 32) * 512; };

  auto gemm = [&](const float* W32, const float* gamma, const float* beta,
                  const float* bias, int O, int C, int Cpad, size_t P,
                  const _Float16* Xin, _Float16* Xout) {
    int Opad = (O + 63) & ~63;
    pn2_conv_w<<<cdiv((size_t)Opad * Cpad, 256), 256, 0, stream>>>(W32, Wh, O, C, Opad, Cpad);
    dim3 gg(cdiv(ptile8(P), 8), (unsigned)(Opad / 64));
    pn2_gemm_f16<<<gg, 256, 0, stream>>>(Wh, Xin, bias, Ybuf, O, Cpad >> 5, (int)P);
    if (gamma) {
      pn2_bn_stats<<<O, 256, 0, stream>>>(Ybuf, (int)P, gamma, beta, scaleB, shiftB);
      if (Xout) {
        int nKcOut = O >> 5;
        if (P % 128 != 0) {  // p-tile padding exists -> pre-zero swizzled buffer
          size_t ze = xfElems(P, O);
          pn2_zero_f16<<<cdiv(ze, 256), 256, 0, stream>>>(Xout, ze);
        }
        size_t tot = (size_t)O * P;
        pn2_bn_relu_f16<<<cdiv(tot, 256), 256, 0, stream>>>(Ybuf, scaleB, shiftB, Xout, P, tot, nKcOut);
      }
    }
  };

  // ---------------- stage 0: transpose points ----------------
  pn2_pts_transpose<<<cdiv((size_t)B * N * 3, 256), 256, 0, stream>>>(points, xyz1, B, N);

  // ---------------- SA1 ----------------
  pn2_fps<16><<<B, 256, 0, stream>>>(xyz1, N, M1, cidx1, nxyz1);
  {  // grouped input: Cin=3 padded to 32 -> zero then scatter
    size_t ze = xfElems(P1, 32);
    pn2_zero_f16<<<cdiv(ze, 256), 256, 0, stream>>>(XfA, ze);
  }
  pn2_group<<<cdiv((size_t)B * M1, 8), 256, 0, stream>>>(
      xyz1, nxyz1, nullptr, XfA, B, N, M1, K1, 0, 32 / 32, R1SQ);
  gemm(SA1W[0], SA1G[0], SA1B[0], nullptr, 64, 3, 32, P1, XfA, XfB);
  gemm(SA1W[1], SA1G[1], SA1B[1], nullptr, 64, 64, 64, P1, XfB, XfA);
  gemm(SA1W[2], SA1G[2], SA1B[2], nullptr, 128, 64, 64, P1, XfA, nullptr);
  pn2_bn_relu_maxpool<<<cdiv((size_t)B * M1 * 128, 256), 256, 0, stream>>>(
      Ybuf, scaleB, shiftB, feat1, 128, B * M1, K1);

  // ---------------- SA2 ----------------
  pn2_fps<2><<<B, 256, 0, stream>>>(nxyz1, M1, M2, cidx2, nxyz2);
  {  // Cin=131 padded to 160 -> zero then scatter
    size_t ze = xfElems(P2, 160);
    pn2_zero_f16<<<cdiv(ze, 256), 256, 0, stream>>>(XfA, ze);
  }
  pn2_group<<<cdiv((size_t)B * M2, 8), 256, 0, stream>>>(
      nxyz1, nxyz2, feat1, XfA, B, M1, M2, K2, 128, 160 / 32, R2SQ);
  gemm(SA2W[0], SA2G[0], SA2B[0], nullptr, 128, 131, 160, P2, XfA, XfB);
  gemm(SA2W[1], SA2G[1], SA2B[1], nullptr, 128, 128, 128, P2, XfB, XfA);
  gemm(SA2W[2], SA2G[2], SA2B[2], nullptr, 256, 128, 128, P2, XfA, nullptr);
  pn2_bn_relu_maxpool<<<cdiv((size_t)B * M2 * 256, 256), 256, 0, stream>>>(
      Ybuf, scaleB, shiftB, feat2, 256, B * M2, K2);

  // ---------------- local stack ----------------
  {  // Cin=259 padded to 288 -> zero then scatter
    size_t ze = xfElems(P3, 288);
    pn2_zero_f16<<<cdiv(ze, 256), 256, 0, stream>>>(XfA, ze);
  }
  pn2_concat_local<<<cdiv((size_t)259 * P3, 256), 256, 0, stream>>>(
      nxyz2, feat2, XfA, (int)P3, 256, 288 / 32);
  gemm(LW[0], LG[0], LB[0], nullptr, 256, 259, 288, P3, XfA, XfB);
  gemm(LW[1], LG[1], LB[1], nullptr, 512, 256, 256, P3, XfB, XfA);
  gemm(LW[2], LG[2], LB[2], nullptr, 1024, 512, 512, P3, XfA, nullptr);
  pn2_bn_relu_maxpool<<<cdiv((size_t)B * 1024, 256), 256, 0, stream>>>(
      Ybuf, scaleB, shiftB, pooled, 1024, B, M2);

  // ---------------- global FC layers (P = 32, padded to 8 p-tiles) --------
  {
    size_t ze = xfElems(PB, 1024);
    pn2_zero_f16<<<cdiv(ze, 256), 256, 0, stream>>>(XfA, ze);
  }
  pn2_pool_to_x<<<cdiv((size_t)B * 1024, 256), 256, 0, stream>>>(pooled, XfA, B, 1024, 1024 / 32);
  gemm(GW[0], GG[0], GBe[0], GBi[0], 512, 1024, 1024, PB, XfA, XfB);
  gemm(GW[1], GG[1], GBe[1], GBi[1], 256, 512, 512, PB, XfB, XfA);

  // ---------------- classifier (O=40 padded to 64 in Wf) ----------------
  gemm(clsW, nullptr, nullptr, clsB, 40, 256, 256, PB, XfA, nullptr);
  pn2_out_write<<<cdiv((size_t)40 * PB, 256), 256, 0, stream>>>(Ybuf, out, 40, (int)PB);
}
